// MultiHeadAttention_24215025615258
// MI455X (gfx1250) — matrix operand order repaired and checked by reading, not compiled, not run on hardware
//
#include <hip/hip_runtime.h>
#include <hip/hip_bf16.h>

#define N_TOK 4096
#define DIM   256
#define NHEAD 8
#define NEGV  (-4294967295.0f)

typedef _Float16 half8  __attribute__((ext_vector_type(8)));
typedef _Float16 half16 __attribute__((ext_vector_type(16)));
typedef float    float8 __attribute__((ext_vector_type(8)));
typedef unsigned int uvec4 __attribute__((ext_vector_type(4)));
typedef int          ivec4 __attribute__((ext_vector_type(4)));
typedef int          ivec8 __attribute__((ext_vector_type(8)));

#if defined(__has_builtin)
#  if __has_builtin(__builtin_amdgcn_tensor_load_to_lds) && __has_builtin(__builtin_amdgcn_s_wait_tensorcnt)
#    define HAVE_TDM 1
#  endif
#endif

static __device__ __forceinline__ float8 zero8() {
  float8 z;
#pragma unroll
  for (int i = 0; i < 8; ++i) z[i] = 0.0f;
  return z;
}

// A-fragment (16x32 f16): lane lo holds row M=lo; K halves hi*16+0..15 (contiguous)
static __device__ __forceinline__ half16 load_a_frag(const _Float16* base, int ld, int lane) {
  const int lo = lane & 15, hi = lane >> 4;
  const half8 x0 = *(const half8*)(base + lo * ld + hi * 16);
  const half8 x1 = *(const half8*)(base + lo * ld + hi * 16 + 8);
  half16 a;
#pragma unroll
  for (int i = 0; i < 8; ++i) { a[i] = x0[i]; a[i + 8] = x1[i]; }
  return a;
}

// B-fragment (32x16 f16): lane lo holds column N=lo; K halves hi*16+0..15 (contiguous)
static __device__ __forceinline__ half16 load_b_frag(const _Float16* base, int ld, int lane) {
  const int lo = lane & 15, hi = lane >> 4;
  return *(const half16*)(base + lo * ld + hi * 16);
}

static __device__ __forceinline__ float8 wmma_f16(half16 a, half16 b, float8 c) {
  return __builtin_amdgcn_wmma_f32_16x16x32_f16(false, a, false, b, (short)0, c, false, false);
}

// Stage a `rows` x `width` f16 tile (row stride == width, rows contiguous) from
// global memory into LDS. CDNA5 path: one Tensor Data Mover descriptor
// (tensor_load_to_lds, tracked by TENSORcnt) issued by wave 0; fallback:
// cooperative copy. Caller must __syncthreads() afterwards.
static __device__ __forceinline__ void stage_rows_to_lds(_Float16* lds, const _Float16* gsrc,
                                                         int rows, int width, int tid) {
#ifdef HAVE_TDM
  if (tid == 0) {
    const unsigned lds_off = (unsigned)(size_t)lds;        // generic LDS addr[31:0] = byte offset
    const unsigned long long ga = (unsigned long long)(size_t)gsrc;
    uvec4 g0;
    g0[0] = 1u;                                            // count=1, user descriptor
    g0[1] = lds_off;                                       // lds_addr
    g0[2] = (unsigned)ga;                                  // global_addr[31:0]
    g0[3] = (unsigned)((ga >> 32) & 0x01FFFFFFull) | (2u << 30); // addr[56:32] | type=2 (image)
    ivec8 g1;
    g1[0] = (int)(1u << 16);                               // data_size=1 -> 2-byte elems; no multicast/pad
    g1[1] = (int)((unsigned)(width & 0xFFFF) << 16);       // tensor_dim0[15:0] (bits 63:48)
    g1[2] = (int)(((unsigned)width >> 16) |                // tensor_dim0[31:16]
                  ((unsigned)(rows & 0xFFFF) << 16));      // tensor_dim1[15:0]
    g1[3] = (int)(((unsigned)rows >> 16) |                 // tensor_dim1[31:16]
                  ((unsigned)(width & 0xFFFF) << 16));     // tile_dim0
    g1[4] = rows & 0xFFFF;                                 // tile_dim1 (tile_dim2=0)
    g1[5] = width;                                         // tensor_dim0_stride[31:0]
    g1[6] = 0;                                             // stride hi | tensor_dim1_stride lo
    g1[7] = 0;
    ivec4 gz;
    gz[0] = gz[1] = gz[2] = gz[3] = 0;
#if __clang_major__ >= 23
    ivec8 gz8;
#pragma unroll
    for (int i = 0; i < 8; ++i) gz8[i] = 0;
    __builtin_amdgcn_tensor_load_to_lds(g0, g1, gz, gz, gz8, 0);
#else
    __builtin_amdgcn_tensor_load_to_lds(g0, g1, gz, gz, 0);
#endif
    __builtin_amdgcn_s_wait_tensorcnt(0);
  }
#else
  const int chunks = rows * width / 8;
  for (int i = tid; i < chunks; i += 128)
    *(half8*)(lds + i * 8) = *(const half8*)(gsrc + (size_t)i * 8);
#endif
}

// ---------------- f32 -> f16 weight conversion ----------------
__global__ void cvt_f32_f16(const float* __restrict__ src, _Float16* __restrict__ dst, int n) {
  int i = blockIdx.x * blockDim.x + threadIdx.x;
  if (i < n) dst[i] = (_Float16)src[i];
}

// ---------------- LayerNorm (one row per block, 256 threads) ----------------
__global__ void __launch_bounds__(DIM) layernorm_f16(const float* __restrict__ x,
                                                     const float* __restrict__ g,
                                                     const float* __restrict__ b,
                                                     _Float16* __restrict__ out) {
  __shared__ float s1[DIM], s2[DIM];
  const int row = blockIdx.x, tid = threadIdx.x;
  float v = x[(size_t)row * DIM + tid];
  s1[tid] = v; s2[tid] = v * v;
  __syncthreads();
  for (int st = DIM / 2; st > 0; st >>= 1) {
    if (tid < st) { s1[tid] += s1[tid + st]; s2[tid] += s2[tid + st]; }
    __syncthreads();
  }
  float mu  = s1[0] * (1.0f / DIM);
  float var = s2[0] * (1.0f / DIM) - mu * mu;
  float r   = rsqrtf(var + 1e-5f);
  out[(size_t)row * DIM + tid] = (_Float16)((v - mu) * r * g[tid] + b[tid]);
}

// ---------------- Q/K projection: C(e-rows, n-cols) = W x_ln^T ----------------
// The 16-row W tile is shared by all 4 waves -> staged once via TDM into LDS.
__global__ void __launch_bounds__(128) qk_proj(const _Float16* __restrict__ Wh,   // [H][D][D]
                                               const _Float16* __restrict__ act,  // [N][D]
                                               const float* __restrict__ bias,    // [H][D]
                                               _Float16* __restrict__ outb,       // [H][N][D]
                                               float scale) {
  extern __shared__ char smem[];
  _Float16* ldsW = (_Float16*)smem;                  // [16][DIM]
  const int tid = threadIdx.x;
  const int lane = tid & 31, wave = tid >> 5;
  const int lo = lane & 15, hi = lane >> 4;
  const int head = blockIdx.z;
  const int e0 = blockIdx.y * 16;
  const int n0 = (blockIdx.x * 4 + wave) * 16;
  const _Float16* W = Wh + (size_t)head * DIM * DIM;

  stage_rows_to_lds(ldsW, W + (size_t)e0 * DIM, 16, DIM, tid);
  __syncthreads();

  float8 c = zero8();
#pragma unroll
  for (int kb = 0; kb < DIM; kb += 32) {
    half16 a = load_a_frag(ldsW + kb, DIM, lane);                      // ds reads
    half16 b = load_b_frag(act + (size_t)n0 * DIM + kb, DIM, lane);    // global reads
    c = wmma_f16(a, b, c);
  }
  const float8 bv = *(const float8*)(bias + head * DIM + e0 + hi * 8);
  half8 o;
#pragma unroll
  for (int r = 0; r < 8; ++r) o[r] = (_Float16)((c[r] + bv[r]) * scale);
  *(half8*)(outb + (size_t)head * N_TOK * DIM + (size_t)(n0 + lo) * DIM + e0 + hi * 8) = o;
}

// ---------------- V projection: C(n-rows, e-cols), stored transposed [H][D][N] ----------------
__global__ void __launch_bounds__(128) v_proj(const _Float16* __restrict__ Wh,
                                              const _Float16* __restrict__ act,
                                              const float* __restrict__ bias,
                                              _Float16* __restrict__ outT /* [H][D][N] */) {
  extern __shared__ char smem[];
  _Float16* ldsW = (_Float16*)smem;                  // [16][DIM]
  const int tid = threadIdx.x;
  const int lane = tid & 31, wave = tid >> 5;
  const int lo = lane & 15, hi = lane >> 4;
  const int head = blockIdx.z;
  const int e0 = blockIdx.y * 16;
  const int n0 = (blockIdx.x * 4 + wave) * 16;
  const _Float16* W = Wh + (size_t)head * DIM * DIM;

  stage_rows_to_lds(ldsW, W + (size_t)e0 * DIM, 16, DIM, tid);
  __syncthreads();

  float8 c = zero8();
#pragma unroll
  for (int kb = 0; kb < DIM; kb += 32) {
    half16 a = load_a_frag(act + (size_t)n0 * DIM + kb, DIM, lane);
    half16 b = load_b_frag(ldsW + kb, DIM, lane);
    c = wmma_f16(a, b, c);
  }
  const float bv = bias[head * DIM + e0 + lo];
  half8 o;
#pragma unroll
  for (int r = 0; r < 8; ++r) o[r] = (_Float16)(c[r] + bv);
  *(half8*)(outT + (size_t)head * DIM * N_TOK + (size_t)(e0 + lo) * N_TOK + n0 + hi * 8) = o;
}

// ---------------- Fused attention: one block = (head, 16 query rows) ----------------
// 320 KB WGP LDS holds the full 16x4096 f32 score stripe + O accumulator + Q tile.
__global__ void __launch_bounds__(128) attn_fused(const _Float16* __restrict__ q,
                                                  const _Float16* __restrict__ k,
                                                  const _Float16* __restrict__ vT,
                                                  const float* __restrict__ spatial,
                                                  const float* __restrict__ edge,
                                                  const int* __restrict__ mask_idx,
                                                  _Float16* __restrict__ cat /* [N][H*D] */) {
  extern __shared__ char smem[];
  float* stripe = (float*)smem;                                   // [4096 keys][16 queries]
  float* Oacc   = (float*)(smem + N_TOK * 16 * 4);                // [16][256]
  float* redb   = (float*)(smem + N_TOK * 16 * 4 + 16 * DIM * 4); // [16*8]
  float* rowmax = redb + 128;                                     // [16]
  float* rowsum = rowmax + 16;                                    // [16]
  _Float16* qtile = (_Float16*)(rowsum + 16);                     // [16][256] f16 (8 KB)

  const int tid = threadIdx.x;
  const int lane = tid & 31, wave = tid >> 5;
  const int lo = lane & 15, hi = lane >> 4;
  const int head = blockIdx.y;
  const int q0 = blockIdx.x * 16;

  const _Float16* qh = q  + (size_t)head * N_TOK * DIM + (size_t)q0 * DIM;
  const _Float16* kh = k  + (size_t)head * N_TOK * DIM;
  const _Float16* vh = vT + (size_t)head * DIM * N_TOK;

  // Stage shared Q tile (16x256, scale folded in) once via TDM
  stage_rows_to_lds(qtile, qh, 16, DIM, tid);
  __syncthreads();

  half16 aq[8];
#pragma unroll
  for (int kk = 0; kk < 8; ++kk) aq[kk] = load_a_frag(qtile + kk * 32, DIM, lane);

  int mq[8];
#pragma unroll
  for (int r = 0; r < 8; ++r) mq[r] = mask_idx[q0 + hi * 8 + r];

  // ---- phase 1: S = QK^T + spatial + edge (masked) -> LDS stripe ----
  for (int kt = wave; kt < N_TOK / 16; kt += 4) {
    const int m0 = kt * 16;
    if (kt + 4 < N_TOK / 16)
      __builtin_prefetch(kh + (size_t)(m0 + 64) * DIM + lane * 16, 0, 0);
    float8 c = zero8();
#pragma unroll
    for (int kk = 0; kk < 8; ++kk) {
      half16 b = load_b_frag(kh + (size_t)m0 * DIM + kk * 32, DIM, lane);
      c = wmma_f16(aq[kk], b, c);
    }
    const int m  = m0 + lo;
    const int mk = mask_idx[m];
    float8 sv;
#pragma unroll
    for (int r = 0; r < 8; ++r) {
      const int n = q0 + hi * 8 + r;
      float s = c[r] + spatial[(size_t)n * N_TOK + m] + edge[(size_t)n * N_TOK + m];
      sv[r] = (mq[r] * mk == 0) ? NEGV : s;
    }
    *(float8*)&stripe[m * 16 + hi * 8] = sv;
  }
  for (int i = tid; i < 16 * DIM; i += 128) Oacc[i] = 0.0f;  // zero O while stripe lands
  __syncthreads();

  // ---- phase 2: exact softmax over each 4096-wide stripe row ----
  const int qr = tid & 15;
  const int seg = tid >> 4;  // 8 segments of 512 keys
  float mx = -3.0e38f;
  for (int mi = seg * 512; mi < (seg + 1) * 512; ++mi)
    mx = fmaxf(mx, stripe[mi * 16 + qr]);
  redb[qr * 8 + seg] = mx;
  __syncthreads();
  if (tid < 16) {
    float m2 = redb[tid * 8];
#pragma unroll
    for (int j = 1; j < 8; ++j) m2 = fmaxf(m2, redb[tid * 8 + j]);
    rowmax[tid] = m2;
  }
  __syncthreads();
  const float M = rowmax[qr];
  float sm = 0.0f;
  for (int mi = seg * 512; mi < (seg + 1) * 512; ++mi) {
    float p = __expf(stripe[mi * 16 + qr] - M);  // masked: exp(-4.3e9) == 0
    stripe[mi * 16 + qr] = p;
    sm += p;
  }
  redb[qr * 8 + seg] = sm;
  __syncthreads();
  if (tid < 16) {
    float s2 = 0.0f;
#pragma unroll
    for (int j = 0; j < 8; ++j) s2 += redb[tid * 8 + j];
    rowsum[tid] = s2;
  }
  __syncthreads();

  // ---- phase 3: O += P * V, P A-frags from LDS, V B-frags contiguous from vT ----
  float8 co[16];
#pragma unroll
  for (int et = 0; et < 16; ++et) co[et] = zero8();
  for (int cc = wave; cc < N_TOK / 32; cc += 4) {
    const int m0 = cc * 32;
    half16 pa;
#pragma unroll
    for (int j = 0; j < 8; ++j) {
      pa[j]     = (_Float16)stripe[(m0 + hi * 16 + j) * 16 + lo];
      pa[j + 8] = (_Float16)stripe[(m0 + hi * 16 + 8 + j) * 16 + lo];
    }
#pragma unroll
    for (int et = 0; et < 16; ++et) {
      half16 b = *(const half16*)(vh + (size_t)(et * 16 + lo) * N_TOK + m0 + hi * 16);
      co[et] = wmma_f16(pa, b, co[et]);
    }
  }
#pragma unroll
  for (int et = 0; et < 16; ++et)
#pragma unroll
    for (int r = 0; r < 8; ++r)
      atomicAdd(&Oacc[(hi * 8 + r) * DIM + et * 16 + lo], co[et][r]);
  __syncthreads();

  // normalize, zero fully-masked query rows, write f16 into concatenated heads
  for (int i = tid; i < 16 * DIM; i += 128) {
    const int qq = i >> 8;
    const int e  = i & 255;
    const float l = rowsum[qq];
    const int mgood = mask_idx[q0 + qq];
    float val = (mgood != 0 && l > 0.0f) ? (Oacc[i] / l) : 0.0f;
    cat[(size_t)(q0 + qq) * (NHEAD * DIM) + head * DIM + e] = (_Float16)val;
  }
}

// ---------------- GEMM + bias + residual (f32 out): out = act @ W^T + bias + res ----------------
// Shared 16-row W tile (16 x K halves) staged via TDM into LDS once per block.
__global__ void __launch_bounds__(128) gemm_bias_res(const _Float16* __restrict__ W,   // [DIM][K]
                                                     const _Float16* __restrict__ act, // [N][K]
                                                     const float* __restrict__ bias,   // [DIM]
                                                     const float* __restrict__ res,    // [N][DIM]
                                                     float* __restrict__ out,          // [N][DIM]
                                                     int K) {
  extern __shared__ char smem[];
  _Float16* ldsW = (_Float16*)smem;                  // [16][K]
  const int tid = threadIdx.x;
  const int lane = tid & 31, wave = tid >> 5;
  const int lo = lane & 15, hi = lane >> 4;
  const int e0 = blockIdx.y * 16;
  const int n0 = (blockIdx.x * 4 + wave) * 16;

  stage_rows_to_lds(ldsW, W + (size_t)e0 * K, 16, K, tid);
  __syncthreads();

  float8 c = zero8();
  for (int kb = 0; kb < K; kb += 32) {
    half16 a = load_a_frag(ldsW + kb, K, lane);
    half16 b = load_b_frag(act + (size_t)n0 * K + kb, K, lane);
    c = wmma_f16(a, b, c);
  }
  const float8 bv = *(const float8*)(bias + e0 + hi * 8);
  const size_t off = (size_t)(n0 + lo) * DIM + e0 + hi * 8;
  const float8 rv = *(const float8*)(res + off);
  float8 o;
#pragma unroll
  for (int r = 0; r < 8; ++r) o[r] = c[r] + bv[r] + rv[r];
  *(float8*)(out + off) = o;
}

extern "C" void kernel_launch(void* const* d_in, const int* in_sizes, int n_in,
                              void* d_out, int out_size, void* d_ws, size_t ws_size,
                              hipStream_t stream) {
  const float* x       = (const float*)d_in[0];
  const int*   mask    = (const int*)d_in[1];
  const float* spatial = (const float*)d_in[2];
  const float* edge    = (const float*)d_in[3];
  const float* ln1_g   = (const float*)d_in[4];
  const float* ln1_b   = (const float*)d_in[5];
  const float* Wq      = (const float*)d_in[6];
  const float* bq      = (const float*)d_in[7];
  const float* Wk      = (const float*)d_in[8];
  const float* bk      = (const float*)d_in[9];
  const float* Wv      = (const float*)d_in[10];
  const float* bv      = (const float*)d_in[11];
  const float* Wo      = (const float*)d_in[12];
  const float* bo      = (const float*)d_in[13];
  const float* ln2_g   = (const float*)d_in[14];
  const float* ln2_b   = (const float*)d_in[15];
  const float* Wff     = (const float*)d_in[16];
  const float* bff     = (const float*)d_in[17];
  float* out = (float*)d_out;

  char* ws = (char*)d_ws;
  size_t off = 0;
  _Float16* xlnH  = (_Float16*)(ws + off); off += (size_t)N_TOK * DIM * 2;            // 2 MB
  _Float16* qH    = (_Float16*)(ws + off); off += (size_t)NHEAD * N_TOK * DIM * 2;    // 16 MB
  _Float16* kH    = (_Float16*)(ws + off); off += (size_t)NHEAD * N_TOK * DIM * 2;    // 16 MB
  _Float16* vTH   = (_Float16*)(ws + off); off += (size_t)NHEAD * DIM * N_TOK * 2;    // 16 MB
  _Float16* catH  = (_Float16*)(ws + off); off += (size_t)N_TOK * NHEAD * DIM * 2;    // 16 MB
  float*    xoutF = (float*)(ws + off);    off += (size_t)N_TOK * DIM * 4;            // 4 MB
  _Float16* xln2H = (_Float16*)(ws + off); off += (size_t)N_TOK * DIM * 2;            // 2 MB
  _Float16* WqH   = (_Float16*)(ws + off); off += (size_t)NHEAD * DIM * DIM * 2;      // 1 MB
  _Float16* WkH   = (_Float16*)(ws + off); off += (size_t)NHEAD * DIM * DIM * 2;      // 1 MB
  _Float16* WvH   = (_Float16*)(ws + off); off += (size_t)NHEAD * DIM * DIM * 2;      // 1 MB
  _Float16* WoH   = (_Float16*)(ws + off); off += (size_t)DIM * NHEAD * DIM * 2;      // 1 MB
  _Float16* WffH  = (_Float16*)(ws + off); off += (size_t)DIM * DIM * 2;              // 128 KB

  const int nWqkv = NHEAD * DIM * DIM;
  const int nWo   = DIM * NHEAD * DIM;
  const int nWff  = DIM * DIM;
  cvt_f32_f16<<<(nWqkv + 255) / 256, 256, 0, stream>>>(Wq,  WqH,  nWqkv);
  cvt_f32_f16<<<(nWqkv + 255) / 256, 256, 0, stream>>>(Wk,  WkH,  nWqkv);
  cvt_f32_f16<<<(nWqkv + 255) / 256, 256, 0, stream>>>(Wv,  WvH,  nWqkv);
  cvt_f32_f16<<<(nWo   + 255) / 256, 256, 0, stream>>>(Wo,  WoH,  nWo);
  cvt_f32_f16<<<(nWff  + 255) / 256, 256, 0, stream>>>(Wff, WffH, nWff);

  layernorm_f16<<<N_TOK, DIM, 0, stream>>>(x, ln1_g, ln1_b, xlnH);

  // QKV projections (scale 1/sqrt(D) folded into q); 8 KB LDS W-tile per block
  dim3 gProj(N_TOK / 64, DIM / 16, NHEAD);
  const size_t projSmem = (size_t)16 * DIM * 2;
  qk_proj<<<gProj, 128, projSmem, stream>>>(WqH, xlnH, bq, qH, 0.0625f);
  qk_proj<<<gProj, 128, projSmem, stream>>>(WkH, xlnH, bk, kH, 1.0f);
  v_proj <<<gProj, 128, projSmem, stream>>>(WvH, xlnH, bv, vTH);

  // Fused attention: 256 KB stripe + 16 KB O + 640 B reductions + 8 KB Q tile
  const size_t attnSmem = (size_t)N_TOK * 16 * 4 + (size_t)16 * DIM * 4 + 160 * 4
                        + (size_t)16 * DIM * 2;
  attn_fused<<<dim3(N_TOK / 16, NHEAD), 128, attnSmem, stream>>>(
      qH, kH, vTH, spatial, edge, mask, catH);

  // output projection + residual (64 KB LDS W-tile)
  gemm_bias_res<<<dim3(N_TOK / 64, DIM / 16), 128, (size_t)16 * NHEAD * DIM * 2, stream>>>(
      WoH, catH, bo, x, xoutF, NHEAD * DIM);

  // LN2 + FF + residual (8 KB LDS W-tile)
  layernorm_f16<<<N_TOK, DIM, 0, stream>>>(xoutF, ln2_g, ln2_b, xln2H);
  gemm_bias_res<<<dim3(N_TOK / 64, DIM / 16), 128, (size_t)16 * DIM * 2, stream>>>(
      WffH, xln2H, bff, xoutF, out, DIM);
}
